// RetrievalFusionModel_70849780515093
// MI455X (gfx1250) — matrix operand order repaired and checked by reading, not compiled, not run on hardware
//
#include <hip/hip_runtime.h>
#include <math.h>

// ---------------- problem constants ----------------
#define B_Q    1024
#define N_DB   100000
#define D_K    1024
#define ES     101
#define TOPK   70
#define NSPLIT 50
#define COLS_SPLIT (N_DB / NSPLIT)   // 2000, divisible by 16
#define MBLK   128                   // queries per block (8 wave row-tiles)
#define KCHUNK 128                   // K staged per LDS buffer
#define KW (KCHUNK + 16)             // padded LDS row pitch (halves)
#define H1 256
#define H2 128
#define INV_TEMP 25.0f               // 1 / 0.04
#define LNEPS 1e-5f

typedef __attribute__((ext_vector_type(16))) __bf16 v16bf;
typedef __attribute__((ext_vector_type(8)))  float  v8f;
typedef __attribute__((ext_vector_type(4)))  unsigned int u32x4;
typedef __attribute__((ext_vector_type(4)))  int i32x4;

union Frag { v16bf v; u32x4 q[2]; };

// ---- async LDS staging (gfx1250 GLOBAL_LOAD_ASYNC_TO_LDS_B128), guarded ----
// ROCm 7.2 clang declares the builtin with strictly-typed pointers:
//   (int4 addrspace(1)*, int4 addrspace(3)*, imm offset, imm cpol)
#if __has_builtin(__builtin_amdgcn_global_load_async_to_lds_b128)
#define ASYNC_LDS 1
typedef __attribute__((address_space(1))) i32x4 gbl_i32x4;
typedef __attribute__((address_space(3))) i32x4 lds_i32x4;
#endif

__device__ __forceinline__ void stage16(unsigned short* ldsDst, const unsigned short* gSrc)
{
#ifdef ASYNC_LDS
    __builtin_amdgcn_global_load_async_to_lds_b128(
        (gbl_i32x4*)gSrc, (lds_i32x4*)ldsDst, /*offset=*/0, /*cpol=*/0);
#else
    *(u32x4*)ldsDst = *(const u32x4*)gSrc;
#endif
}

__device__ __forceinline__ void stage_wait()
{
#ifdef ASYNC_LDS
#if __has_builtin(__builtin_amdgcn_s_wait_asynccnt)
    __builtin_amdgcn_s_wait_asynccnt(0);
#else
    asm volatile("s_wait_asynccnt 0x0" ::: "memory");
#endif
#endif
}

// ---------------- row L2-normalize + fp32->bf16 ----------------
__global__ void k_rownorm_bf16(const float* __restrict__ src,
                               unsigned short* __restrict__ dst)
{
    __shared__ float red[256];
    const int tid = threadIdx.x;
    const size_t row = blockIdx.x;
    const float* s = src + row * D_K;
    float ss = 0.f;
#pragma unroll
    for (int i = 0; i < D_K / 256; ++i) { float v = s[tid + i * 256]; ss += v * v; }
    red[tid] = ss; __syncthreads();
    for (int off = 128; off > 0; off >>= 1) {
        if (tid < off) red[tid] += red[tid + off];
        __syncthreads();
    }
    const float inv = 1.f / fmaxf(sqrtf(red[0]), 1e-12f);
    unsigned short* d = dst + row * D_K;
#pragma unroll
    for (int i = 0; i < D_K / 256; ++i) {
        float v = s[tid + i * 256] * inv;
        unsigned int u = __float_as_uint(v);
        d[tid + i * 256] = (unsigned short)((u + 0x7FFFu + ((u >> 16) & 1u)) >> 16);
    }
}

// ---------------- fused bf16-WMMA GEMM + per-query top-70 ----------------
// grid: (NSPLIT, B_Q/MBLK), 256 threads. Each wave owns a 16-row tile; the
// whole block shares each 16-column B tile staged (double-buffered) in LDS.
__global__ void __launch_bounds__(256) k_sims_topk(
    const unsigned short* __restrict__ qn,
    const unsigned short* __restrict__ dbn,
    float* __restrict__ partV, int* __restrict__ partI)
{
    extern __shared__ unsigned char smem[];
    unsigned short* sB   = (unsigned short*)smem;                       // [2][16][KW]
    float* sTile = (float*)(smem + (size_t)2 * 16 * KW * sizeof(unsigned short)); // [8][16][17]
    float* sTopV = sTile + 8 * 16 * 17;                                 // [MBLK][TOPK]
    int*   sTopI = (int*)(sTopV + MBLK * TOPK);                         // [MBLK][TOPK]

    const int tid  = threadIdx.x;
    const int wave = tid >> 5;
    const int lane = tid & 31;
    const int colBase = blockIdx.x * COLS_SPLIT;
    const int qbase   = blockIdx.y * MBLK;

    for (int i = tid; i < MBLK * TOPK; i += 256) { sTopV[i] = -3.0e38f; sTopI[i] = 0; }
    __syncthreads();

    // CDNA5 16-bit A 16x32 layout: lane holds row m=lane&15,
    // K halves {0..7,16..23} (lanes 0-15) / {8..15,24..31} (lanes 16-31).
    const unsigned short* aRow =
        qn + (size_t)(qbase + wave * 16 + (lane & 15)) * D_K + ((lane >> 4) << 3);

    // staging map: thread -> (col 0..15, 8-half chunk of K) for b128 transfers
    const int sCol = tid >> 4;
    const int sK   = (tid & 15) << 3;

    for (int it = 0; it < COLS_SPLIT / 16; ++it) {
        const int n0 = colBase + it * 16;
        const unsigned short* bRow = dbn + (size_t)(n0 + sCol) * D_K + sK;

        // stage K-chunk 0 into buffer 0 (16B per thread, coalesced)
        stage16(&sB[sCol * KW + sK], bRow);
        if (it + 1 < COLS_SPLIT / 16)
            __builtin_prefetch(dbn + (size_t)(n0 + 16 + sCol) * D_K + sK, 0, 1);
        stage_wait();
        __syncthreads();

        v8f acc = {};
#pragma unroll 1
        for (int c = 0; c < D_K / KCHUNK; ++c) {
            const int buf = c & 1;
            if (c + 1 < D_K / KCHUNK)
                stage16(&sB[(buf ^ 1) * 16 * KW + sCol * KW + sK],
                        bRow + (c + 1) * KCHUNK);

            // B 32x16 layout: lane holds col n=lane&15,
            // K halves {0..7,16..23} (lanes 0-15) / {8..15,24..31} (lanes 16-31).
            const unsigned short* bb =
                &sB[buf * 16 * KW + (lane & 15) * KW + ((lane >> 4) << 3)];
            const unsigned short* ab = aRow + c * KCHUNK;
#pragma unroll
            for (int ks = 0; ks < KCHUNK; ks += 32) {
                Frag A, Bm;
                A.q[0]  = *(const u32x4*)(ab + ks);
                A.q[1]  = *(const u32x4*)(ab + ks + 16);
                Bm.q[0] = *(const u32x4*)(bb + ks);
                Bm.q[1] = *(const u32x4*)(bb + ks + 16);
                acc = __builtin_amdgcn_wmma_f32_16x16x32_bf16(
                    false, A.v, false, Bm.v, (short)0, acc, false, false);
            }
            stage_wait();        // next chunk's async transfer has landed
            __syncthreads();
        }

        // dump C (v: lanes0-15 M=v,N=lane; lanes16-31 M=v+8,N=lane-16)
        {
            const int r0 = (lane >> 4) << 3, cc = lane & 15;
#pragma unroll
            for (int j = 0; j < 8; ++j)
                sTile[(wave * 16 + r0 + j) * 17 + cc] = acc[j];
        }
        __syncthreads();

        // one thread per block-local query maintains a sorted top-70 list
        if (tid < MBLK) {
            const int w = tid >> 4, r = tid & 15;
            float* tv = sTopV + tid * TOPK;
            int*   ti = sTopI + tid * TOPK;
            float th = tv[TOPK - 1];
            for (int ccc = 0; ccc < 16; ++ccc) {
                const float v = sTile[(w * 16 + r) * 17 + ccc];
                if (v > th) {
                    int pos = TOPK - 1;
                    while (pos > 0 && tv[pos - 1] < v) {
                        tv[pos] = tv[pos - 1]; ti[pos] = ti[pos - 1]; --pos;
                    }
                    tv[pos] = v; ti[pos] = n0 + ccc;
                    th = tv[TOPK - 1];
                }
            }
        }
        __syncthreads();
    }

    for (int i = tid; i < MBLK * TOPK; i += 256) {
        const int q = i / TOPK, k = i - q * TOPK;
        const size_t o = ((size_t)(qbase + q) * NSPLIT + blockIdx.x) * TOPK + k;
        partV[o] = sTopV[q * TOPK + k];
        partI[o] = sTopI[q * TOPK + k];
    }
}

// ---------------- merge NSPLIT partial lists + softmax(1/TEMP) ----------------
__global__ void k_merge_softmax(const float* __restrict__ partV,
                                const int* __restrict__ partI,
                                float* __restrict__ wts, int* __restrict__ fidx)
{
    extern __shared__ unsigned char smem[];
    float* mv = (float*)smem + (size_t)threadIdx.x * TOPK;
    int*   mi = (int*)((float*)smem + (size_t)blockDim.x * TOPK) + (size_t)threadIdx.x * TOPK;
    const int q = blockIdx.x * blockDim.x + threadIdx.x;
    const float* pv = partV + (size_t)q * NSPLIT * TOPK;
    const int*   pi = partI + (size_t)q * NSPLIT * TOPK;

    for (int k = 0; k < TOPK; ++k) { mv[k] = pv[k]; mi[k] = pi[k]; }
    for (int s = 1; s < NSPLIT; ++s) {
        const float* sv = pv + s * TOPK; const int* si = pi + s * TOPK;
        for (int k = 0; k < TOPK; ++k) {
            const float v = sv[k];
            if (v <= mv[TOPK - 1]) break;   // split lists are sorted descending
            int pos = TOPK - 1;
            while (pos > 0 && mv[pos - 1] < v) {
                mv[pos] = mv[pos - 1]; mi[pos] = mi[pos - 1]; --pos;
            }
            mv[pos] = v; mi[pos] = si[k];
        }
    }
    const float m = mv[0];
    float sum = 0.f;
    for (int k = 0; k < TOPK; ++k) sum += __expf((mv[k] - m) * INV_TEMP);
    const float inv = 1.f / sum;
    for (int k = 0; k < TOPK; ++k) {
        wts[(size_t)q * TOPK + k]  = __expf((mv[k] - m) * INV_TEMP) * inv;
        fidx[(size_t)q * TOPK + k] = mi[k];
    }
}

// ---------------- weighted gather of the 101 selector columns ----------------
__global__ void k_gather(const float* __restrict__ wts, const int* __restrict__ fidx,
                         const float* __restrict__ es_db, const int* __restrict__ sel,
                         float* __restrict__ esr, float* __restrict__ out)
{
    __shared__ float w[TOPK];
    __shared__ int   id[TOPK];
    const int tid = threadIdx.x, q = blockIdx.x;
    if (tid < TOPK) { w[tid] = wts[(size_t)q * TOPK + tid]; id[tid] = fidx[(size_t)q * TOPK + tid]; }
    __syncthreads();
    if (tid < ES) {
        const int col = sel[tid];
        float acc = 0.f;
        for (int k = 0; k < TOPK; ++k)
            acc = fmaf(w[k], es_db[(size_t)id[k] * D_K + col], acc);
        esr[(size_t)q * ES + tid] = acc;
        out[(size_t)B_Q * ES + (size_t)q * ES + tid] = acc;   // es_retrieved_101
    }
}

// ---------------- tiny MLP head (fp32 VALU; negligible runtime) ----------------
__device__ __forceinline__ float block_sum256(float v, float* red, int tid)
{
    red[tid] = v; __syncthreads();
    for (int off = 128; off > 0; off >>= 1) {
        if (tid < off) red[tid] += red[tid + off];
        __syncthreads();
    }
    const float r = red[0]; __syncthreads();
    return r;
}

__global__ void __launch_bounds__(256) k_mlp(
    const float* __restrict__ en, const float* __restrict__ esr,
    const float* __restrict__ W1, const float* __restrict__ b1,
    const float* __restrict__ g1, const float* __restrict__ be1,
    const float* __restrict__ W2, const float* __restrict__ b2,
    const float* __restrict__ g2, const float* __restrict__ be2,
    const float* __restrict__ W3, const float* __restrict__ b3,
    float* __restrict__ out)
{
    __shared__ float feats[D_K + ES];
    __shared__ float h1[H1];
    __shared__ float h2[H2];
    __shared__ float red[256];
    const int tid = threadIdx.x;
    const size_t q = blockIdx.x;

#pragma unroll
    for (int i = 0; i < D_K / 256; ++i)
        feats[tid + i * 256] = en[q * D_K + tid + i * 256];
    if (tid < ES) feats[D_K + tid] = esr[q * ES + tid];
    __syncthreads();

    // layer 1: 1125 -> 256, layernorm, gelu(exact)
    float s = b1[tid];
    for (int i = 0; i < D_K + ES; ++i) s = fmaf(feats[i], W1[(size_t)i * H1 + tid], s);
    float mu  = block_sum256(s, red, tid) * (1.f / H1);
    float d   = s - mu;
    float var = block_sum256(d * d, red, tid) * (1.f / H1);
    float x   = d * rsqrtf(var + LNEPS) * g1[tid] + be1[tid];
    h1[tid]   = 0.5f * x * (1.f + erff(x * 0.70710678118f));
    __syncthreads();

    // layer 2: 256 -> 128, layernorm, gelu
    float s2 = 0.f;
    if (tid < H2) {
        s2 = b2[tid];
        for (int i = 0; i < H1; ++i) s2 = fmaf(h1[i], W2[(size_t)i * H2 + tid], s2);
    }
    float mu2  = block_sum256(tid < H2 ? s2 : 0.f, red, tid) * (1.f / H2);
    float d2   = s2 - mu2;
    float var2 = block_sum256(tid < H2 ? d2 * d2 : 0.f, red, tid) * (1.f / H2);
    if (tid < H2) {
        float x2 = d2 * rsqrtf(var2 + LNEPS) * g2[tid] + be2[tid];
        h2[tid]  = 0.5f * x2 * (1.f + erff(x2 * 0.70710678118f));
    }
    __syncthreads();

    // layer 3: 128 -> 101, residual + outputs
    if (tid < ES) {
        float s3 = b3[tid];
        for (int i = 0; i < H2; ++i) s3 = fmaf(h2[i], W3[(size_t)i * ES + tid], s3);
        const float er = feats[D_K + tid];
        out[q * ES + tid] = er + s3;                        // es_pred_101
        out[(size_t)2 * B_Q * ES + q * ES + tid] = s3;      // delta
    }
}

// ---------------- launch ----------------
extern "C" void kernel_launch(void* const* d_in, const int* in_sizes, int n_in,
                              void* d_out, int out_size, void* d_ws, size_t ws_size,
                              hipStream_t stream)
{
    (void)in_sizes; (void)n_in; (void)out_size; (void)ws_size;
    const float* en    = (const float*)d_in[0];
    const float* en_db = (const float*)d_in[1];
    const float* es_db = (const float*)d_in[2];
    const int*   sel   = (const int*)d_in[3];
    const float* W1  = (const float*)d_in[4];  const float* b1  = (const float*)d_in[5];
    const float* g1  = (const float*)d_in[6];  const float* be1 = (const float*)d_in[7];
    const float* W2  = (const float*)d_in[8];  const float* b2  = (const float*)d_in[9];
    const float* g2  = (const float*)d_in[10]; const float* be2 = (const float*)d_in[11];
    const float* W3  = (const float*)d_in[12]; const float* b3  = (const float*)d_in[13];
    float* out = (float*)d_out;

    unsigned char* ws = (unsigned char*)d_ws;
    size_t off = 0;
    auto alloc = [&](size_t bytes) -> unsigned char* {
        off = (off + 255) & ~(size_t)255;
        unsigned char* p = ws + off;
        off += bytes;
        return p;
    };
    unsigned short* qn    = (unsigned short*)alloc((size_t)B_Q * D_K * 2);
    unsigned short* dbn   = (unsigned short*)alloc((size_t)N_DB * D_K * 2);
    float*          partV = (float*)alloc((size_t)B_Q * NSPLIT * TOPK * 4);
    int*            partI = (int*)  alloc((size_t)B_Q * NSPLIT * TOPK * 4);
    float*          wts   = (float*)alloc((size_t)B_Q * TOPK * 4);
    int*            fidx  = (int*)  alloc((size_t)B_Q * TOPK * 4);
    float*          esr   = (float*)alloc((size_t)B_Q * ES * 4);

    k_rownorm_bf16<<<B_Q,  256, 0, stream>>>(en, qn);
    k_rownorm_bf16<<<N_DB, 256, 0, stream>>>(en_db, dbn);

    const size_t smem1 = (size_t)2 * 16 * KW * 2        // B double buffer
                       + (size_t)8 * 16 * 17 * 4        // C dump tiles
                       + (size_t)MBLK * TOPK * (4 + 4); // top-k val+idx
    k_sims_topk<<<dim3(NSPLIT, B_Q / MBLK), 256, smem1, stream>>>(qn, dbn, partV, partI);

    const size_t smem2 = (size_t)128 * TOPK * (4 + 4);
    k_merge_softmax<<<dim3(B_Q / 128), 128, smem2, stream>>>(partV, partI, wts, fidx);

    k_gather<<<B_Q, 128, 0, stream>>>(wts, fidx, es_db, sel, esr, out);
    k_mlp<<<B_Q, 256, 0, stream>>>(en, esr, W1, b1, g1, be1, W2, b2, g2, be2, W3, b3, out);
}
